// Multi_Head_Attention_Fusion_37168646979688
// MI455X (gfx1250) — compile-verified, matrix operand order checked by reading, not run on hardware
//
#include <hip/hip_runtime.h>
#include <hip/hip_bf16.h>

// Problem constants
#define Bc 2
#define Sc 384
#define Hc 768
#define NHc 12
#define DPHc 64
#define SCALEc 0.125f   // 1/sqrt(64)

typedef __bf16 bf16_t;
typedef __attribute__((ext_vector_type(16))) __bf16 v16bf;
typedef __attribute__((ext_vector_type(8)))  float  v8f;
typedef __attribute__((ext_vector_type(4)))  unsigned int v4u;
typedef __attribute__((ext_vector_type(8)))  int v8i;
typedef __attribute__((ext_vector_type(4)))  int v4i;

// ---------------------------------------------------------------------------
// Tensor Data Mover: 1-D strip load global -> LDS.
// D# per CDNA5 ISA:
//   group0: [1:0]=count=1, [63:32]=lds_addr, [120:64]=global_addr, [127:126]=type=2
//   group1: [17:16]=data_size(3 => 8B), [79:48]=tensor_dim0, [111:80]=tensor_dim1,
//           [127:112]=tile_dim0, [143:128]=tile_dim1(0=unused),
//           [207:160]=tensor_dim0_stride
//   groups 2/3 (+ trailing group): zero (<=2-D tensor)
// n8 = number of 8-byte elements. Toolchain uses the 6-arg builtin form.
// ---------------------------------------------------------------------------
__device__ __forceinline__ void tdm_load_1d(unsigned lds_off,
                                            unsigned long long gaddr,
                                            unsigned n8) {
  v4u g0;
  g0[0] = 1u;                                                   // count = 1
  g0[1] = lds_off;                                              // lds_addr
  g0[2] = (unsigned)(gaddr & 0xffffffffull);                    // ga[31:0]
  g0[3] = (unsigned)((gaddr >> 32) & 0x1ffffffull) | (2u << 30);// ga[56:32]|type=2
  v8i g1 = {};
  g1[0] = (int)(3u << 16);                                      // data_size = 8B
  g1[1] = (int)((n8 & 0xffffu) << 16);                          // tensor_dim0 lo
  g1[2] = (int)(((n8 >> 16) & 0xffffu) | (1u << 16));           // dim0 hi | dim1=1
  g1[3] = (int)((n8 & 0xffffu) << 16);                          // tile_dim0
  g1[4] = 0;                                                    // tile_dim1/2 unused
  g1[5] = (int)n8;                                              // dim0_stride lo
  g1[6] = 0;
  g1[7] = 0;
  v4i z4 = {};
  v8i z8 = {};
  __builtin_amdgcn_tensor_load_to_lds(g0, g1, z4, z4, z8, 0);
}

// ---------------------------------------------------------------------------
// WMMA fragment loaders (V_WMMA_F32_16X16X32_BF16 operand layouts, ISA 7.12.2)
// A (16x32, row-major per lane): lane<16 -> row=lane,   K = {k0..k0+7, k0+16..k0+23}
//                                lane>=16 -> row=lane-16,K = {k0+8..k0+15, k0+24..k0+31}
// B (32x16) mirrors A with N in place of M (column of B = row of B^T).
// ---------------------------------------------------------------------------
__device__ __forceinline__ v16bf frag_row_f32(const float* rowbase, int k0, int lane) {
  const float* p = rowbase + k0 + ((lane & 16) ? 8 : 0);
  v16bf r;
#pragma unroll
  for (int t = 0; t < 8; ++t) { r[t] = (bf16_t)p[t]; r[8 + t] = (bf16_t)p[16 + t]; }
  return r;
}
__device__ __forceinline__ v16bf frag_row_bf16(const bf16_t* rowbase, int k0, int lane) {
  const bf16_t* p = rowbase + k0 + ((lane & 16) ? 8 : 0);
  v16bf r;
#pragma unroll
  for (int t = 0; t < 8; ++t) { r[t] = p[t]; r[8 + t] = p[16 + t]; }
  return r;
}
__device__ __forceinline__ v16bf frag_col_f32(const float* colbase, int ldb, int k0, int lane) {
  const int off = k0 + ((lane & 16) ? 8 : 0);
  v16bf r;
#pragma unroll
  for (int t = 0; t < 8; ++t) {
    r[t]     = (bf16_t)colbase[(long)(off + t) * ldb];
    r[8 + t] = (bf16_t)colbase[(long)(off + 16 + t) * ldb];
  }
  return r;
}

// ---------------------------------------------------------------------------
// Generic bf16 WMMA GEMM:  C[M,N] = A[M,K] * B + bias, fp32 in, TC out.
// BT=1: B stored [N,K] (torch Linear weight, x@W.T). BT=0: B stored [K,N].
// grid.x = M/16 (A strip staged in LDS as bf16), grid.z = batch,
// block = 256 (8 wave32), each wave owns 16x16 C tiles strided by 8.
// ---------------------------------------------------------------------------
template <int BT, typename TC>
__global__ void gemm_wmma_kernel(const float* __restrict__ A, long sAb, int lda,
                                 const float* __restrict__ Bm, long sBb, int ldb,
                                 const float* __restrict__ bias,
                                 TC* __restrict__ C, long sCb, int ldc,
                                 int N, int K, float alpha) {
  __shared__ __align__(16) bf16_t sA[16 * 768];
  const int z = blockIdx.z;
  A  += (long)z * sAb;
  Bm += (long)z * sBb;
  C  += (long)z * sCb;
  const int m0  = blockIdx.x * 16;
  const int tid = threadIdx.x;

  // Cooperative A strip load [16 x K] with fp32 -> bf16 conversion
  for (int idx = tid; idx < 16 * K; idx += 256) {
    const int r  = idx / K;
    const int kk = idx - r * K;
    sA[r * K + kk] = (bf16_t)A[(long)(m0 + r) * lda + kk];
  }
  __syncthreads();

  const int lane  = tid & 31;
  const int wave  = tid >> 5;
  const int row16 = lane & 15;
  const bf16_t* arow = sA + row16 * K;
  const int nt = N >> 4;

  for (int ct = wave; ct < nt; ct += 8) {
    const int n0 = ct * 16;
    v8f acc = {};
    for (int k0 = 0; k0 < K; k0 += 32) {
      v16bf a = frag_row_bf16(arow, k0, lane);
      v16bf b;
      if (BT) b = frag_row_f32(Bm + (long)(n0 + row16) * ldb, k0, lane);
      else    b = frag_col_f32(Bm + (n0 + row16), ldb, k0, lane);
      acc = __builtin_amdgcn_wmma_f32_16x16x32_bf16(false, a, false, b,
                                                    (short)0, acc, false, false);
    }
    const int   col   = n0 + row16;
    const float bvv   = bias ? bias[col] : 0.0f;
    const int   mbase = m0 + ((lane >> 4) << 3);
#pragma unroll
    for (int r = 0; r < 8; ++r)
      C[(long)(mbase + r) * ldc + col] = (TC)(acc[r] * alpha + bvv);
  }
}

// qv[i] = q[i] + rowvec[i % H]   (broadcast of flat [NH*DPH] bias over rows)
__global__ void add_rowvec_kernel(const float* __restrict__ x,
                                  const float* __restrict__ rv,
                                  float* __restrict__ o, int n) {
  const int i = blockIdx.x * 256 + threadIdx.x;
  if (i < n) o[i] = x[i] + rv[i % Hc];
}

// Fill Aext second half (block-diagonal q+u for the AC term), zero padded head
// rows 12..15, and compute the br constant cst[bi,h] = (q+v)_h . br_h
__global__ void prep_aext_kernel(const float* __restrict__ qproj,
                                 const float* __restrict__ qv,
                                 const float* __restrict__ u,
                                 const float* __restrict__ br,
                                 bf16_t* __restrict__ Aext,
                                 float* __restrict__ cst) {
  const int bi  = blockIdx.x;
  const int tid = threadIdx.x;
  bf16_t* Ab = Aext + (long)bi * 16 * 1536;
  // cols 768..1535: Ab[h][768+c] = (c>>6 == h) ? q[bi,c] + u[c] : 0
  for (int idx = tid; idx < 16 * Hc; idx += 256) {
    const int h = idx / Hc;
    const int c = idx - h * Hc;
    float val = 0.0f;
    if (h < NHc && (c >> 6) == h) val = qproj[(long)bi * Hc + c] + u[c];
    Ab[h * 1536 + 768 + c] = (bf16_t)val;
  }
  // zero rows 12..15 of the first half (GEMM only wrote heads 0..11)
  for (int idx = tid; idx < 4 * Hc; idx += 256) {
    const int h = NHc + idx / Hc;
    const int c = idx - (h - NHc) * Hc;
    Ab[h * 1536 + c] = (bf16_t)0.0f;
  }
  if (tid < 16) {
    float s = 0.0f;
    if (tid < NHc)
      for (int d = 0; d < DPHc; ++d)
        s += qv[(long)bi * Hc + tid * DPHc + d] * br[tid * DPHc + d];
    cst[bi * 16 + tid] = s;
  }
}

// One workgroup per (b,i): scores[b,h,i,:] = Aext[bi] (16x1536 bf16) times
// [rpe[b,i,:,:] | kproj[b,:,:]]^T (1536x384), epilogue adds cst and scales.
// Aext strip staged into LDS via the Tensor Data Mover (wave 0 issues the DMA).
__global__ void rel_scores_kernel(const bf16_t* __restrict__ Aext,
                                  const float* __restrict__ rpe,
                                  const float* __restrict__ kproj,
                                  const float* __restrict__ cst,
                                  float* __restrict__ scores) {
  __shared__ __align__(16) bf16_t sA[16 * 1536];   // 48 KB
  const int bi  = blockIdx.x;
  const int b   = bi / Sc;
  const int i   = bi - b * Sc;
  const int tid = threadIdx.x;

  if (tid < 32) {  // wave 0: one TDM descriptor moves the whole 48 KB strip
    tdm_load_1d((unsigned)(size_t)(void*)sA,
                (unsigned long long)(size_t)(const void*)(Aext + (long)bi * 16 * 1536),
                (16u * 1536u * 2u) / 8u);
    __builtin_amdgcn_s_wait_tensorcnt(0);
  }
  __syncthreads();

  const int lane  = tid & 31;
  const int wave  = tid >> 5;
  const int row16 = lane & 15;
  const bf16_t* arow = sA + row16 * 1536;

  for (int ct = wave; ct < Sc / 16; ct += 8) {
    const int j = ct * 16 + row16;
    const float* rrow = rpe   + (((long)bi * Sc) + j) * Hc;  // rpe[b,i,j,:]
    const float* krow = kproj + (((long)b  * Sc) + j) * Hc;  // k[b,j,:]

    // Prefetch next j-tile's rpe row (emits global_prefetch_b8)
    const int jn = (ct + 8) * 16 + row16;
    if (jn < Sc) {
      const float* nrow = rpe + (((long)bi * Sc) + jn) * Hc;
      __builtin_prefetch(nrow, 0, 1);
      __builtin_prefetch(nrow + 384, 0, 1);
    }

    v8f acc = {};
    for (int t = 0; t < 48; ++t) {
      const int k0 = t * 32;
      v16bf a  = frag_row_bf16(arow, k0, lane);
      v16bf bf = (t < 24) ? frag_row_f32(rrow, k0, lane)
                          : frag_row_f32(krow, k0 - 768, lane);
      acc = __builtin_amdgcn_wmma_f32_16x16x32_bf16(false, a, false, bf,
                                                    (short)0, acc, false, false);
    }
    const int hbase = (lane >> 4) << 3;
#pragma unroll
    for (int r = 0; r < 8; ++r) {
      const int h = hbase + r;
      if (h < NHc) {
        const float s = (acc[r] + cst[bi * 16 + h]) * SCALEc;
        scores[((((long)b * NHc + h) * Sc) + i) * Sc + (ct * 16 + row16)] = s;
      }
    }
  }
}

// Row softmax over j with the additive mask; 128 threads, 3 elems each.
__global__ void softmax_kernel(float* __restrict__ scores,
                               const int* __restrict__ mask) {
  const long row = blockIdx.x;                 // (b*NH + h)*S + i
  const int  b   = (int)(row / ((long)NHc * Sc));
  float* p = scores + row * Sc;
  const int* mr = mask + b * Sc;
  __shared__ float red[128];
  const int tid = threadIdx.x;
  float x[3];
  float mx = -3.0e38f;
#pragma unroll
  for (int t = 0; t < 3; ++t) {
    const int j = tid + 128 * t;
    const float e = (1.0f - (float)mr[j]) * -1.0e18f;
    x[t] = p[j] + e;
    mx = fmaxf(mx, x[t]);
  }
  red[tid] = mx; __syncthreads();
  for (int st = 64; st > 0; st >>= 1) {
    if (tid < st) red[tid] = fmaxf(red[tid], red[tid + st]);
    __syncthreads();
  }
  mx = red[0]; __syncthreads();
  float s = 0.0f;
#pragma unroll
  for (int t = 0; t < 3; ++t) { x[t] = __expf(x[t] - mx); s += x[t]; }
  red[tid] = s; __syncthreads();
  for (int st = 64; st > 0; st >>= 1) {
    if (tid < st) red[tid] += red[tid + st];
    __syncthreads();
  }
  const float inv = 1.0f / red[0];
#pragma unroll
  for (int t = 0; t < 3; ++t) p[tid + 128 * t] = x[t] * inv;
}

// [B,S,NH,D] -> [B,NH,S,D]
__global__ void permute_v_kernel(const float* __restrict__ vin, float* __restrict__ o) {
  const long idx = (long)blockIdx.x * 256 + threadIdx.x;    // B*S*H elements
  const int d = (int)(idx & 63);
  long r = idx >> 6;
  const int h = (int)(r % NHc); r /= NHc;
  const int s = (int)(r % Sc);
  const int b = (int)(r / Sc);
  o[((((long)b * NHc + h) * Sc) + s) * DPHc + d] = vin[idx];
}
// [B,NH,S,D] -> [B,S,NH*D]
__global__ void unpermute_ctx_kernel(const float* __restrict__ cin, float* __restrict__ o) {
  const long idx = (long)blockIdx.x * 256 + threadIdx.x;
  const int d = (int)(idx & 63);
  long r = idx >> 6;
  const int s = (int)(r % Sc); r /= Sc;
  const int h = (int)(r % NHc);
  const int b = (int)(r / NHc);
  o[(((long)b * Sc + s) * NHc + h) * DPHc + d] = cin[idx];
}

// out = LayerNorm(attn + hidden) * g + b   (256 threads, 3 elems each)
__global__ void layernorm_kernel(const float* __restrict__ attn,
                                 const float* __restrict__ hidden,
                                 const float* __restrict__ g,
                                 const float* __restrict__ be,
                                 float* __restrict__ out) {
  const int row = blockIdx.x;
  const int tid = threadIdx.x;
  const float* a = attn   + (long)row * Hc;
  const float* h = hidden + (long)row * Hc;
  __shared__ float red[256];
  float x[3];
  float s = 0.0f;
#pragma unroll
  for (int t = 0; t < 3; ++t) { const int c = tid + 256 * t; x[t] = a[c] + h[c]; s += x[t]; }
  red[tid] = s; __syncthreads();
  for (int st = 128; st > 0; st >>= 1) { if (tid < st) red[tid] += red[tid + st]; __syncthreads(); }
  const float mean = red[0] * (1.0f / Hc); __syncthreads();
  s = 0.0f;
#pragma unroll
  for (int t = 0; t < 3; ++t) { const float d = x[t] - mean; s += d * d; }
  red[tid] = s; __syncthreads();
  for (int st = 128; st > 0; st >>= 1) { if (tid < st) red[tid] += red[tid + st]; __syncthreads(); }
  const float rstd = rsqrtf(red[0] * (1.0f / Hc) + 1e-6f);
#pragma unroll
  for (int t = 0; t < 3; ++t) {
    const int c = tid + 256 * t;
    out[(long)row * Hc + c] = (x[t] - mean) * rstd * g[c] + be[c];
  }
}

extern "C" void kernel_launch(void* const* d_in, const int* in_sizes, int n_in,
                              void* d_out, int out_size, void* d_ws, size_t ws_size,
                              hipStream_t stream) {
  const float* hidden = (const float*)d_in[0];
  const int*   mask   = (const int*)  d_in[1];
  const float* rpe    = (const float*)d_in[2];
  const float* Wq = (const float*)d_in[3];  const float* bq = (const float*)d_in[4];
  const float* Wk = (const float*)d_in[5];  const float* bk = (const float*)d_in[6];
  const float* Wv = (const float*)d_in[7];  const float* bv = (const float*)d_in[8];
  const float* Wr = (const float*)d_in[9];  const float* br = (const float*)d_in[10];
  const float* u  = (const float*)d_in[11]; const float* v  = (const float*)d_in[12];
  const float* Wf = (const float*)d_in[13]; const float* bfw = (const float*)d_in[14];
  const float* lng = (const float*)d_in[15]; const float* lnb = (const float*)d_in[16];
  float* out = (float*)d_out;

  char* ws = (char*)d_ws;
  const long NROW = (long)Bc * Sc;              // 768
  const long MATF = NROW * Hc * sizeof(float);  // 2359296 B
  float*  qproj = (float*)(ws);
  float*  kproj = (float*)(ws + 1 * MATF);
  float*  vproj = (float*)(ws + 2 * MATF);
  float*  qv    = (float*)(ws + 3 * MATF);
  bf16_t* aext  = (bf16_t*)(ws + 4 * MATF);                 // [768][16][1536] bf16
  char*   p     = ws + 4 * MATF + NROW * 16 * 1536 * sizeof(bf16_t);
  float*  cst   = (float*)p;            p += NROW * 16 * sizeof(float);
  float*  scores= (float*)p;            p += (long)Bc * NHc * Sc * Sc * sizeof(float);
  float*  vperm = (float*)p;            p += MATF;
  float*  ctxp  = (float*)p;            p += MATF;
  float*  ctx   = (float*)p;            p += MATF;
  float*  attn  = (float*)p;

  const dim3 blk(256);

  // 1) q/k/v projections: [768,768] x W.T + b   (WMMA bf16)
  gemm_wmma_kernel<1, float><<<dim3(48, 1, 1), blk, 0, stream>>>(
      hidden, 0, Hc, Wq, 0, Hc, bq, qproj, 0, Hc, Hc, Hc, 1.0f);
  gemm_wmma_kernel<1, float><<<dim3(48, 1, 1), blk, 0, stream>>>(
      hidden, 0, Hc, Wk, 0, Hc, bk, kproj, 0, Hc, Hc, Hc, 1.0f);
  gemm_wmma_kernel<1, float><<<dim3(48, 1, 1), blk, 0, stream>>>(
      hidden, 0, Hc, Wv, 0, Hc, bv, vproj, 0, Hc, Hc, Hc, 1.0f);

  // 2) qv = q + broadcast(v_bias)
  add_rowvec_kernel<<<dim3((int)((NROW * Hc + 255) / 256)), blk, 0, stream>>>(
      qproj, v, qv, (int)(NROW * Hc));

  // 3) w_qv[bi,h,:] = qv[bi, h-slice] @ Wr[h-slice, :]  (batched over 12 heads,
  //    K=64, bf16 output straight into Aext first half)
  gemm_wmma_kernel<0, bf16_t><<<dim3(48, 1, NHc), blk, 0, stream>>>(
      qv, 64, Hc, Wr, (long)64 * Hc, Hc, nullptr,
      aext, 1536, 16 * 1536, Hc, 64, 1.0f);

  // 4) Aext second half (block-diagonal q+u), pad rows, br constants
  prep_aext_kernel<<<dim3((int)NROW), blk, 0, stream>>>(qproj, qv, u, br, aext, cst);

  // 5) scores = (AC + Bp + Dp) * SCALE via one fused WMMA GEMM per (b,i)
  rel_scores_kernel<<<dim3((int)NROW), blk, 0, stream>>>(aext, rpe, kproj, cst, scores);

  // 6) masked softmax over j
  softmax_kernel<<<dim3(Bc * NHc * Sc), dim3(128), 0, stream>>>(scores, mask);

  // 7) ctx = probs @ val   (batched WMMA, B*NH batches of [384x384]x[384x64])
  permute_v_kernel<<<dim3((int)((NROW * Hc) / 256)), blk, 0, stream>>>(vproj, vperm);
  gemm_wmma_kernel<0, float><<<dim3(Sc / 16, 1, Bc * NHc), blk, 0, stream>>>(
      scores, (long)Sc * Sc, Sc, vperm, (long)Sc * DPHc, DPHc, nullptr,
      ctxp, (long)Sc * DPHc, DPHc, DPHc, Sc, 1.0f);
  unpermute_ctx_kernel<<<dim3((int)((NROW * Hc) / 256)), blk, 0, stream>>>(ctxp, ctx);

  // 8) out projection + residual layernorm
  gemm_wmma_kernel<1, float><<<dim3(48, 1, 1), blk, 0, stream>>>(
      ctx, 0, Hc, Wf, 0, Hc, bfw, attn, 0, Hc, Hc, Hc, 1.0f);
  layernorm_kernel<<<dim3((int)NROW), blk, 0, stream>>>(attn, hidden, lng, lnb, out);
}
